// VectorQuantizer_24206435680854
// MI455X (gfx1250) — matrix operand order repaired and checked by reading, not compiled, not run on hardware
//
#include <hip/hip_runtime.h>
#include <hip/hip_bf16.h>

// ---------------------------------------------------------------------------
// VQ (vector-quantizer) forward for MI455X / gfx1250.
// All GEMMs use v_wmma_f32_16x16x32_bf16 with hi/lo split-compensation
// (A*B ~= Ahi*Bhi + Ahi*Blo + Alo*Bhi) for ~fp32 accuracy at matrix-core rate.
// Pure-copy LDS staging uses GLOBAL_LOAD_ASYNC_TO_LDS_B128 (ASYNCcnt), and the
// dominant dist GEMM double-buffers the codebook stage: s_wait_asynccnt 0x10
// retires the previous stage while the next one is still streaming.
// ---------------------------------------------------------------------------

typedef __attribute__((ext_vector_type(16))) __bf16 v16bf;
typedef __attribute__((ext_vector_type(8)))  __bf16 v8bf;
typedef __attribute__((ext_vector_type(8)))  float  v8f;

#define IN_CH   512
#define EMB_NUM 8192
#define EMB_CH  256
#define MTOT    16384   // 4 * 4096 positions

__device__ __forceinline__ v8f wmma_bf16(v16bf a, v16bf b, v8f c) {
    // 8 args: (neg_a, A, neg_b, B, c_mod, C, reuse_a, reuse_b)
    return __builtin_amdgcn_wmma_f32_16x16x32_bf16(false, a, false, b,
                                                   (short)0, c, false, false);
}

// Assemble a 16-element bf16 fragment from two 8-element (16B) chunks.
__device__ __forceinline__ v16bf frag2(const __bf16* p0, const __bf16* p1) {
    v8bf x = *(const v8bf*)p0;
    v8bf y = *(const v8bf*)p1;
    v16bf r;
#pragma unroll
    for (int i = 0; i < 8; ++i) { r[i] = x[i]; r[i + 8] = y[i]; }
    return r;
}

__device__ __forceinline__ void split_bf16(float f, __bf16& hi, __bf16& lo) {
    hi = (__bf16)f;
    lo = (__bf16)(f - (float)hi);
}

// Async DMA of one 16-byte chunk global -> LDS (tracked by ASYNCcnt).
// lds_off = low 32 bits of the flat LDS address (HW adds LDS_BASE).
__device__ __forceinline__ void async_copy16(unsigned lds_off, const void* g) {
    asm volatile("global_load_async_to_lds_b128 %0, %1, off"
                 :: "v"(lds_off), "v"((unsigned long long)g) : "memory");
}
__device__ __forceinline__ void wait_async0() {
    asm volatile("s_wait_asynccnt 0x0" ::: "memory");
}
__device__ __forceinline__ void wait_async16() {
    asm volatile("s_wait_asynccnt 0x10" ::: "memory");
}
__device__ __forceinline__ unsigned lds_off32(const void* p) {
    return (unsigned)(unsigned long long)p;
}

// ---------------------------------------------------------------------------
// Prep: codebook -> bf16 hi/lo + row norms ||v||^2
// ---------------------------------------------------------------------------
__global__ void k_prep_cb(const float* __restrict__ cb,
                          __bf16* __restrict__ cbh, __bf16* __restrict__ cbl,
                          float* __restrict__ cbn) {
    __shared__ float red[256];
    const int s = blockIdx.x;      // code row
    const int t = threadIdx.x;     // 256 threads == EMB_CH
    const float v = cb[(size_t)s * EMB_CH + t];
    __bf16 h, l; split_bf16(v, h, l);
    cbh[(size_t)s * EMB_CH + t] = h;
    cbl[(size_t)s * EMB_CH + t] = l;
    red[t] = v * v;
    __syncthreads();
#pragma unroll
    for (int o = 128; o > 0; o >>= 1) {
        if (t < o) red[t] += red[t + o];
        __syncthreads();
    }
    if (t == 0) cbn[s] = red[0];
}

// Prep: W_in / W_out -> bf16 hi/lo, zero the loss accumulator.
__global__ void k_prep_w(const float* __restrict__ Wi, const float* __restrict__ Wo,
                         __bf16* __restrict__ Wih, __bf16* __restrict__ Wil,
                         __bf16* __restrict__ Woh, __bf16* __restrict__ Wol,
                         float* __restrict__ lossacc) {
    const int i = blockIdx.x * blockDim.x + threadIdx.x;
    if (i == 0) *lossacc = 0.f;
    if (i < IN_CH * EMB_CH) {
        __bf16 h, l;
        float a = Wi[i]; split_bf16(a, h, l); Wih[i] = h; Wil[i] = l;
        float b = Wo[i]; split_bf16(b, h, l); Woh[i] = h; Wol[i] = l;
    }
}

// ---------------------------------------------------------------------------
// z_e[16384,256] = Z[16384,512] @ W_in^T   (W_in is [256,512] row-major)
// Block: 128 threads = 4 waves, M-tile 64, K staged in chunks of 32.
// B tiles (pure bf16 copy) are staged with async-to-LDS DMA.
// ---------------------------------------------------------------------------
__global__ __launch_bounds__(128) void k_gemm_ze(const float* __restrict__ Z,
                                                 const __bf16* __restrict__ Wih,
                                                 const __bf16* __restrict__ Wil,
                                                 float* __restrict__ ze) {
    __shared__ __bf16 Ah[64 * 32], Al[64 * 32];         //  4 KB each
    __shared__ __bf16 Bh[EMB_CH * 32], Bl[EMB_CH * 32]; // 16 KB each
    const int tid = threadIdx.x, lane = tid & 31, w = tid >> 5;
    const int bm = blockIdx.x * 64;
    const int mb = w * 16;
    const unsigned bh_off = lds_off32(Bh), bl_off = lds_off32(Bl);

    v8f acc[16] = {};   // 16 N-tiles of 16x16 fp32

    for (int kc0 = 0; kc0 < IN_CH; kc0 += 32) {
        // Stage B via async DMA: 256 rows x 32 K of bf16 hi/lo.
        // 1024 16B chunks per array; chunk c: row = c>>2, part = c&3.
#pragma unroll
        for (int i = 0; i < 8; ++i) {
            const int c = tid + i * 128;
            const int row = c >> 2, part = c & 3;
            const unsigned loff = (unsigned)(row * 64 + part * 16);
            const size_t goff = (size_t)row * IN_CH * 2 + (size_t)kc0 * 2 + part * 16;
            async_copy16(bh_off + loff, (const char*)Wih + goff);
            async_copy16(bl_off + loff, (const char*)Wil + goff);
        }
        // Stage A: 64x32 fp32 -> bf16 hi/lo (512 float4, 4 per thread)
#pragma unroll
        for (int i = 0; i < 4; ++i) {
            const int f4 = tid + i * 128;
            const int row = f4 >> 3;             // 8 float4 per row
            const int c4 = (f4 & 7) * 4;
            const float4 v = *(const float4*)&Z[(size_t)(bm + row) * IN_CH + kc0 + c4];
            __bf16 h, l;
            split_bf16(v.x, h, l); Ah[row * 32 + c4 + 0] = h; Al[row * 32 + c4 + 0] = l;
            split_bf16(v.y, h, l); Ah[row * 32 + c4 + 1] = h; Al[row * 32 + c4 + 1] = l;
            split_bf16(v.z, h, l); Ah[row * 32 + c4 + 2] = h; Al[row * 32 + c4 + 2] = l;
            split_bf16(v.w, h, l); Ah[row * 32 + c4 + 3] = h; Al[row * 32 + c4 + 3] = l;
        }
        wait_async0();
        __syncthreads();

        // A fragments (K=32 step): lane half selects contiguous K 0..15 / 16..31
        const __bf16* ap = &Ah[(mb + (lane & 15)) * 32 + ((lane >> 4) * 16)];
        const __bf16* lp = &Al[(mb + (lane & 15)) * 32 + ((lane >> 4) * 16)];
        const v16bf ah = frag2(ap, ap + 8);
        const v16bf al = frag2(lp, lp + 8);
#pragma unroll
        for (int nt = 0; nt < 16; ++nt) {
            const __bf16* bp = &Bh[(nt * 16 + (lane & 15)) * 32 + (lane >> 4) * 16];
            const __bf16* bq = &Bl[(nt * 16 + (lane & 15)) * 32 + (lane >> 4) * 16];
            const v16bf bh = frag2(bp, bp + 8);
            const v16bf bl = frag2(bq, bq + 8);
            acc[nt] = wmma_bf16(ah, bh, acc[nt]);
            acc[nt] = wmma_bf16(ah, bl, acc[nt]);
            acc[nt] = wmma_bf16(al, bh, acc[nt]);
        }
        __syncthreads();
    }
    // C layout: element r -> M = r + 8*(lane>=16), N = lane&15
#pragma unroll
    for (int nt = 0; nt < 16; ++nt)
#pragma unroll
        for (int r = 0; r < 8; ++r)
            ze[(size_t)(bm + mb + (lane >> 4) * 8 + r) * EMB_CH + nt * 16 + (lane & 15)] = acc[nt][r];
}

// ---------------------------------------------------------------------------
// Fused dist + argmin: for each row m, argmin_s (||v_s||^2 - 2 * z_e[m].v_s)
// A fragments (full K=256) live in registers for the entire 68.7 GFLOP GEMM.
// The 64 KB z_e staging buffer is re-purposed as a DOUBLE-BUFFERED codebook
// stage (32 codes / 16 KB per array per buffer) filled by async-to-LDS DMA:
// while buffer k is consumed by 4 waves of WMMAs, buffer k^1 streams in.
// ---------------------------------------------------------------------------
__global__ __launch_bounds__(128) void k_dist_argmin(const float* __restrict__ ze,
                                                     const __bf16* __restrict__ cbh,
                                                     const __bf16* __restrict__ cbl,
                                                     const float* __restrict__ cbn,
                                                     int* __restrict__ idx_i,
                                                     float* __restrict__ idx_f) {
    __shared__ __bf16 smem[2 * 64 * 256];          // 64 KB, dual-purpose
    __bf16* Zh = smem;                             // phase 1: z_e hi
    __bf16* Zl = smem + 64 * 256;                  // phase 1: z_e lo
    const int tid = threadIdx.x, lane = tid & 31, w = tid >> 5;
    const int bm = blockIdx.x * 64;

    // Phase 1: stage z_e tile (64x256 fp32 -> hi/lo): 4096 float4, 32/thread
#pragma unroll
    for (int i = 0; i < 32; ++i) {
        const int f4 = tid + i * 128;
        const int row = f4 >> 6;
        const int c4 = (f4 & 63) * 4;
        const float4 v = *(const float4*)&ze[(size_t)(bm + row) * EMB_CH + c4];
        __bf16 h, l;
        split_bf16(v.x, h, l); Zh[row * 256 + c4 + 0] = h; Zl[row * 256 + c4 + 0] = l;
        split_bf16(v.y, h, l); Zh[row * 256 + c4 + 1] = h; Zl[row * 256 + c4 + 1] = l;
        split_bf16(v.z, h, l); Zh[row * 256 + c4 + 2] = h; Zl[row * 256 + c4 + 2] = l;
        split_bf16(v.w, h, l); Zh[row * 256 + c4 + 3] = h; Zl[row * 256 + c4 + 3] = l;
    }
    __syncthreads();

    const int mb = w * 16;
    v16bf ah[8], al[8];   // full K=256 of A fragments in registers
#pragma unroll
    for (int kc = 0; kc < 8; ++kc) {
        const __bf16* p = &Zh[(mb + (lane & 15)) * 256 + kc * 32 + (lane >> 4) * 16];
        const __bf16* q = &Zl[(mb + (lane & 15)) * 256 + kc * 32 + (lane >> 4) * 16];
        ah[kc] = frag2(p, p + 8);
        al[kc] = frag2(q, q + 8);
    }
    __syncthreads();   // everyone done reading z_e; smem now free for phase 2

    // Phase 2: double-buffered codebook stage.
    // Buffer k (k=0,1): bytes [k*32768, k*32768+32768): Bh at +0, Bl at +16384.
    const unsigned b0 = lds_off32(smem);
    auto issue_stage = [&](int sb, int k) {
        const char* gh = (const char*)cbh + (size_t)sb * EMB_CH * 2;
        const char* gl = (const char*)cbl + (size_t)sb * EMB_CH * 2;
        const unsigned bh = b0 + (unsigned)k * 32768u;
        const unsigned bl = bh + 16384u;
#pragma unroll
        for (int i = 0; i < 8; ++i) {
            const unsigned c = (unsigned)(tid + i * 128) * 16u;  // byte offset
            async_copy16(bh + c, gh + c);
            async_copy16(bl + c, gl + c);
        }
    };

    float rmin[8]; int ridx[8];
#pragma unroll
    for (int r = 0; r < 8; ++r) { rmin[r] = 3.4e38f; ridx[r] = 0; }

    issue_stage(0, 0);   // prologue: fill buffer 0
    for (int sb = 0, it = 0; sb < EMB_NUM; sb += 32, ++it) {
        const int k = it & 1;
        if (sb + 32 < EMB_NUM) {
            issue_stage(sb + 32, k ^ 1);  // stream next stage into other buffer
            wait_async16();               // <=16 outstanding => current stage done
        } else {
            wait_async0();
        }
        __syncthreads();

        const __bf16* BhK = smem + (size_t)k * 16384;  // elems (32768 B / 2)
        const __bf16* BlK = BhK + 8192;
#pragma unroll
        for (int j = 0; j < 2; ++j) {
            const int s0 = sb + j * 16 + (lane & 15);  // code for this lane's column
            const float c = cbn[s0];
            v8f acc = {};
            const __bf16* bh0 = &BhK[(j * 16 + (lane & 15)) * 256 + (lane >> 4) * 16];
            const __bf16* bl0 = &BlK[(j * 16 + (lane & 15)) * 256 + (lane >> 4) * 16];
#pragma unroll
            for (int kc = 0; kc < 8; ++kc) {
                const v16bf bh = frag2(bh0 + kc * 32, bh0 + kc * 32 + 8);
                const v16bf bl = frag2(bl0 + kc * 32, bl0 + kc * 32 + 8);
                acc = wmma_bf16(ah[kc], bh, acc);
                acc = wmma_bf16(ah[kc], bl, acc);
                acc = wmma_bf16(al[kc], bh, acc);
            }
#pragma unroll
            for (int r = 0; r < 8; ++r) {
                const float d = c - 2.f * acc[r];
                if (d < rmin[r]) { rmin[r] = d; ridx[r] = s0; }  // first occurrence wins
            }
        }
        __syncthreads();   // all waves done with buffer k before stage sb+64 rewrites it
    }

    // min-reduce across the 16 lanes of each half (xor masks keep bit 4 intact)
#pragma unroll
    for (int m = 1; m < 16; m <<= 1) {
#pragma unroll
        for (int r = 0; r < 8; ++r) {
            const float ov = __shfl_xor(rmin[r], m, 32);
            const int   oi = __shfl_xor(ridx[r], m, 32);
            if (ov < rmin[r] || (ov == rmin[r] && oi < ridx[r])) { rmin[r] = ov; ridx[r] = oi; }
        }
    }
    if ((lane & 15) == 0) {
#pragma unroll
        for (int r = 0; r < 8; ++r) {
            const int m = bm + mb + (lane >> 4) * 8 + r;
            idx_i[m] = ridx[r];
            idx_f[m] = (float)ridx[r];
        }
    }
}

// ---------------------------------------------------------------------------
// Fused gather + loss + out GEMM: z_q = codebook[idx]; loss += (z_q - z_e)^2;
// out[16384,512] = z_q @ W_out^T  (W_out is [512,256] row-major)
// ---------------------------------------------------------------------------
__global__ __launch_bounds__(128) void k_out(const float* __restrict__ ze,
                                             const float* __restrict__ cb,
                                             const int* __restrict__ idx_i,
                                             const __bf16* __restrict__ Woh,
                                             const __bf16* __restrict__ Wol,
                                             float* __restrict__ out,
                                             float* __restrict__ lossacc) {
    __shared__ __bf16 Qh[64 * 256], Ql[64 * 256];  // 32 KB each
    const int tid = threadIdx.x, lane = tid & 31, w = tid >> 5;
    const int bm = blockIdx.x * 64;

    float lsum = 0.f;
    for (int row = 0; row < 64; ++row) {
        const int s = idx_i[bm + row];             // uniform per row -> scalar load
#pragma unroll
        for (int j = 0; j < 2; ++j) {
            const int col = tid + j * 128;
            const float q = cb[(size_t)s * EMB_CH + col];
            const float z = ze[(size_t)(bm + row) * EMB_CH + col];
            const float d = q - z;
            lsum += d * d;
            __bf16 h, l; split_bf16(q, h, l);
            Qh[row * 256 + col] = h; Ql[row * 256 + col] = l;
        }
    }
    // wave-level loss reduction, one atomic per wave
#pragma unroll
    for (int m = 1; m < 32; m <<= 1) lsum += __shfl_xor(lsum, m, 32);
    if (lane == 0) atomicAdd(lossacc, lsum);
    __syncthreads();

    const int mb = w * 16;
    v16bf ah[8], al[8];
#pragma unroll
    for (int kc = 0; kc < 8; ++kc) {
        const __bf16* p = &Qh[(mb + (lane & 15)) * 256 + kc * 32 + (lane >> 4) * 16];
        const __bf16* q = &Ql[(mb + (lane & 15)) * 256 + kc * 32 + (lane >> 4) * 16];
        ah[kc] = frag2(p, p + 8);
        al[kc] = frag2(q, q + 8);
    }
    for (int nt = 0; nt < IN_CH / 16; ++nt) {       // 32 N-tiles
        v8f acc = {};
        const __bf16* bh0 = Woh + (size_t)(nt * 16 + (lane & 15)) * EMB_CH + (lane >> 4) * 16;
        const __bf16* bl0 = Wol + (size_t)(nt * 16 + (lane & 15)) * EMB_CH + (lane >> 4) * 16;
#pragma unroll
        for (int kc = 0; kc < 8; ++kc) {
            const v16bf bh = frag2(bh0 + kc * 32, bh0 + kc * 32 + 8);
            const v16bf bl = frag2(bl0 + kc * 32, bl0 + kc * 32 + 8);
            acc = wmma_bf16(ah[kc], bh, acc);
            acc = wmma_bf16(ah[kc], bl, acc);
            acc = wmma_bf16(al[kc], bh, acc);
        }
#pragma unroll
        for (int r = 0; r < 8; ++r)
            out[(size_t)(bm + mb + (lane >> 4) * 8 + r) * IN_CH + nt * 16 + (lane & 15)] = acc[r];
    }
}

__global__ void k_finalize(const float* __restrict__ lossacc, float* __restrict__ loss_out) {
    *loss_out = 2.f * (*lossacc) / (float)((size_t)MTOT * EMB_CH);
}

// ---------------------------------------------------------------------------
extern "C" void kernel_launch(void* const* d_in, const int* in_sizes, int n_in,
                              void* d_out, int out_size, void* d_ws, size_t ws_size,
                              hipStream_t stream) {
    const float* Z  = (const float*)d_in[0];  // [4,4096,512]
    const float* cb = (const float*)d_in[1];  // [8192,256]
    const float* Wi = (const float*)d_in[2];  // [256,512]
    const float* Wo = (const float*)d_in[3];  // [512,256]

    float* out      = (float*)d_out;                       // [16384,512]
    float* idx_f    = out + (size_t)MTOT * IN_CH;          // [16384]
    float* loss_out = idx_f + MTOT;                        // [1]

    // Workspace carve-out (~26.3 MB)
    char* ws = (char*)d_ws;
    float*  ze   = (float*)ws;                 ws += (size_t)MTOT * EMB_CH * sizeof(float);
    __bf16* cbh  = (__bf16*)ws;                ws += (size_t)EMB_NUM * EMB_CH * sizeof(__bf16);
    __bf16* cbl  = (__bf16*)ws;                ws += (size_t)EMB_NUM * EMB_CH * sizeof(__bf16);
    float*  cbn  = (float*)ws;                 ws += (size_t)EMB_NUM * sizeof(float);
    __bf16* Wih  = (__bf16*)ws;                ws += (size_t)IN_CH * EMB_CH * sizeof(__bf16);
    __bf16* Wil  = (__bf16*)ws;                ws += (size_t)IN_CH * EMB_CH * sizeof(__bf16);
    __bf16* Woh  = (__bf16*)ws;                ws += (size_t)IN_CH * EMB_CH * sizeof(__bf16);
    __bf16* Wol  = (__bf16*)ws;                ws += (size_t)IN_CH * EMB_CH * sizeof(__bf16);
    int*    idxi = (int*)ws;                   ws += (size_t)MTOT * sizeof(int);
    float*  lacc = (float*)ws;                 ws += sizeof(float);

    k_prep_cb<<<EMB_NUM, EMB_CH, 0, stream>>>(cb, cbh, cbl, cbn);
    k_prep_w<<<(IN_CH * EMB_CH + 255) / 256, 256, 0, stream>>>(Wi, Wo, Wih, Wil, Woh, Wol, lacc);
    k_gemm_ze<<<MTOT / 64, 128, 0, stream>>>(Z, Wih, Wil, ze);
    k_dist_argmin<<<MTOT / 64, 128, 0, stream>>>(ze, cbh, cbl, cbn, idxi, idx_f);
    k_out<<<MTOT / 64, 128, 0, stream>>>(ze, cb, idxi, Woh, Wol, out, lacc);
    k_finalize<<<1, 1, 0, stream>>>(lacc, loss_out);
}
